// Lru_79010218377320
// MI455X (gfx1250) — compile-verified, matrix operand order checked by reading, not run on hardware
//
#include <hip/hip_runtime.h>
#include <hip/hip_bf16.h>
#include <math.h>
#include <stdint.h>

// ---------- types ----------
typedef __attribute__((ext_vector_type(4)))  __bf16 v4bf;
typedef __attribute__((ext_vector_type(8)))  __bf16 v8bf;
typedef __attribute__((ext_vector_type(16))) __bf16 v16bf;
typedef __attribute__((ext_vector_type(8)))  float  v8f;

// ---------- problem constants ----------
#define SEQ_LEN 4096
#define BATCH   4
#define DIM     1024
#define MROWS   (SEQ_LEN * BATCH)   // 16384
#define CHANNELS (BATCH * DIM)      // 4096
#define CHUNK   64
#define NCHUNK  (SEQ_LEN / CHUNK)   // 64

// ---------- GEMM tile config ----------
#define BM 128
#define BN 128
#define BK 32
#define LDT 40   // LDS pitch in bf16 elems (80B, multiple of 16B)

// ---------------------------------------------------------------------------
// f32 -> bf16 pre-convert (vectorized float4 -> 4x bf16)
// ---------------------------------------------------------------------------
__global__ __launch_bounds__(256)
void cvt_f32_bf16(const float* __restrict__ src, __bf16* __restrict__ dst, int n4)
{
    const int i = blockIdx.x * blockDim.x + threadIdx.x;
    if (i < n4) {
        const float4 v = ((const float4*)src)[i];
        v4bf p;
        p[0] = (__bf16)v.x; p[1] = (__bf16)v.y; p[2] = (__bf16)v.z; p[3] = (__bf16)v.w;
        ((v4bf*)dst)[i] = p;
    }
}

// ---------------------------------------------------------------------------
// bf16 WMMA GEMM: C[m,n] = sum_k A[m,k] * B[n,k]
//   A: (M x K) bf16, row stride ldA.  B: (N x K) bf16, row stride K.
// Tiles staged via GLOBAL_LOAD_ASYNC_TO_LDS_B128, double-buffered; the K-loop
// is peeled so the steady state has an unconditional prefetch + asynccnt wait.
// mode 0: val = (C + bias[n]) * exp(-exp(gamma_log[n>>1]));
//         n even -> outR[m*(N/2) + (n>>1)], n odd -> outI[...]
// mode 1: val =  C + bias[n];  outR[m*N + n]
// ---------------------------------------------------------------------------
__global__ __launch_bounds__(256)
void lru_gemm_bf16(const __bf16* __restrict__ A, int ldA,
                   const __bf16* __restrict__ B, int K, int N,
                   const float* __restrict__ bias, const float* __restrict__ gamma_log,
                   float* __restrict__ outR, float* __restrict__ outI, int mode)
{
    __shared__ __attribute__((aligned(16))) __bf16 As[2][BM * LDT];
    __shared__ __attribute__((aligned(16))) __bf16 Bs[2][BN * LDT];

    const int tid   = threadIdx.x;
    const int lane  = tid & 31;
    const int wid   = tid >> 5;
    const int waveM = wid & 3;       // 4 waves along M, 32 rows each
    const int waveN = wid >> 2;      // 2 waves along N, 64 cols each
    const int l16   = lane & 15;
    const int khalf = lane >> 4;

    const int blockN = blockIdx.x * BN;
    const int blockM = blockIdx.y * BM;

    // per-thread async-copy coordinates: 2 b128 loads for A, 2 for B per tile
    const int idx0 = tid * 8;                // bf16 element index, pass 0
    const int idx1 = (256 + tid) * 8;        // pass 1
    const int ra0 = idx0 >> 5, ca0 = idx0 & 31;
    const int ra1 = idx1 >> 5, ca1 = idx1 & 31;

    // global source pointers, strided by BK each tile
    const __bf16* ga0 = A + (size_t)(blockM + ra0) * ldA + ca0;
    const __bf16* ga1 = A + (size_t)(blockM + ra1) * ldA + ca1;
    const __bf16* gb0 = B + (size_t)(blockN + ra0) * K   + ca0;
    const __bf16* gb1 = B + (size_t)(blockN + ra1) * K   + ca1;

    // LDS destination offsets (buffer 0); buffer 1 = +LDS_STRIDE bytes
    const uint32_t la0 = (uint32_t)(uintptr_t)&As[0][ra0 * LDT + ca0];
    const uint32_t la1 = (uint32_t)(uintptr_t)&As[0][ra1 * LDT + ca1];
    const uint32_t lb0 = (uint32_t)(uintptr_t)&Bs[0][ra0 * LDT + ca0];
    const uint32_t lb1 = (uint32_t)(uintptr_t)&Bs[0][ra1 * LDT + ca1];
    const uint32_t LBUF = (uint32_t)(BM * LDT * sizeof(__bf16));   // bytes per buffer

    v8f acc[2][4];
#pragma unroll
    for (int mt = 0; mt < 2; ++mt)
#pragma unroll
        for (int nt = 0; nt < 4; ++nt)
            acc[mt][nt] = (v8f){0.f, 0.f, 0.f, 0.f, 0.f, 0.f, 0.f, 0.f};

#define ISSUE_TILE(kb, p)                                                        \
    do {                                                                         \
        const uint32_t boff = (uint32_t)(p) * LBUF;                              \
        asm volatile("global_load_async_to_lds_b128 %0, %1, off"                 \
                     :: "v"(la0 + boff), "v"(ga0 + (kb)) : "memory");            \
        asm volatile("global_load_async_to_lds_b128 %0, %1, off"                 \
                     :: "v"(la1 + boff), "v"(ga1 + (kb)) : "memory");            \
        asm volatile("global_load_async_to_lds_b128 %0, %1, off"                 \
                     :: "v"(lb0 + boff), "v"(gb0 + (kb)) : "memory");            \
        asm volatile("global_load_async_to_lds_b128 %0, %1, off"                 \
                     :: "v"(lb1 + boff), "v"(gb1 + (kb)) : "memory");            \
    } while (0)

#define COMPUTE_TILE(p)                                                          \
    do {                                                                         \
        const __bf16* curA = As[(p)];                                            \
        const __bf16* curB = Bs[(p)];                                            \
        v16bf afrag[2], bfrag[4];                                                \
        _Pragma("unroll")                                                        \
        for (int mt = 0; mt < 2; ++mt) {                                         \
            const int row = waveM * 32 + mt * 16 + l16;                          \
            const v8bf lo = *(const v8bf*)(&curA[row * LDT + khalf * 8]);        \
            const v8bf hi = *(const v8bf*)(&curA[row * LDT + 16 + khalf * 8]);   \
            afrag[mt] = __builtin_shufflevector(lo, hi,                          \
                0,1,2,3,4,5,6,7,8,9,10,11,12,13,14,15);                          \
        }                                                                        \
        _Pragma("unroll")                                                        \
        for (int nt = 0; nt < 4; ++nt) {                                         \
            const int col = waveN * 64 + nt * 16 + l16;                          \
            const v8bf lo = *(const v8bf*)(&curB[col * LDT + khalf * 8]);        \
            const v8bf hi = *(const v8bf*)(&curB[col * LDT + 16 + khalf * 8]);   \
            bfrag[nt] = __builtin_shufflevector(lo, hi,                          \
                0,1,2,3,4,5,6,7,8,9,10,11,12,13,14,15);                          \
        }                                                                        \
        _Pragma("unroll")                                                        \
        for (int mt = 0; mt < 2; ++mt)                                           \
            _Pragma("unroll")                                                    \
            for (int nt = 0; nt < 4; ++nt)                                       \
                acc[mt][nt] = __builtin_amdgcn_wmma_f32_16x16x32_bf16(           \
                    false, afrag[mt], false, bfrag[nt],                          \
                    (short)0, acc[mt][nt], false, false);                        \
    } while (0)

    ISSUE_TILE(0, 0);

    int parity = 0;
    // steady state: always prefetch next tile (K/BK >= 2 for both GEMMs)
    for (int kb = 0; kb < K - BK; kb += BK) {
        ISSUE_TILE(kb + BK, parity ^ 1);
        // 4 current-tile copies are older than the 4 just issued; in-order
        // completion => asynccnt<=4 means the current tile is resident
        asm volatile("s_wait_asynccnt 0x4" ::: "memory");
        __syncthreads();
        COMPUTE_TILE(parity);
        __syncthreads();
        parity ^= 1;
    }
    // tail: last tile, no prefetch
    asm volatile("s_wait_asynccnt 0x0" ::: "memory");
    __syncthreads();
    COMPUTE_TILE(parity);

#undef ISSUE_TILE
#undef COMPUTE_TILE

    // ---- epilogue ----
#pragma unroll
    for (int mt = 0; mt < 2; ++mt) {
#pragma unroll
        for (int nt = 0; nt < 4; ++nt) {
            const int nn = blockN + waveN * 64 + nt * 16 + l16;
            const int mbase = blockM + waveM * 32 + mt * 16 + khalf * 8;
            const float bv = bias[nn];
            if (mode == 0) {
                const int j = nn >> 1;
                const float g = __expf(-__expf(gamma_log[j]));
                float* dst = ((nn & 1) ? outI : outR) + (size_t)mbase * DIM + j;
#pragma unroll
                for (int v = 0; v < 8; ++v)
                    dst[(size_t)v * DIM] = (acc[mt][nt][v] + bv) * g;
            } else {
                float* dst = outR + (size_t)mbase * N + nn;
#pragma unroll
                for (int v = 0; v < 8; ++v)
                    dst[(size_t)v * N] = acc[mt][nt][v] + bv;
            }
        }
    }
}

// ---------- scan kernels ----------
__device__ __forceinline__ void channel_lambda(const float* nu_log, const float* theta_log,
                                               int j, float& lr, float& li)
{
    const float nu = __expf(nu_log[j]);
    const float th = __expf(theta_log[j]);
    lr = nu * __cosf(th);
    li = nu * __sinf(th);
}

// (a) per-chunk local scan, in place over u (-> h_local); carry per chunk
__global__ __launch_bounds__(256)
void lru_chunk_scan(float* __restrict__ uR, float* __restrict__ uI,
                    float* __restrict__ carR, float* __restrict__ carI,
                    const float* __restrict__ nu_log, const float* __restrict__ theta_log)
{
    const int ch = blockIdx.x * blockDim.x + threadIdx.x;   // 0..4095
    const int c  = blockIdx.y;                              // chunk
    const int j  = ch & (DIM - 1);
    float lr, li; channel_lambda(nu_log, theta_log, j, lr, li);

    float hr = 0.f, hi = 0.f;
    size_t idx = (size_t)c * CHUNK * CHANNELS + ch;
#pragma unroll 4
    for (int t = 0; t < CHUNK; ++t) {
        const float ur = uR[idx], ui = uI[idx];
        const float nhr = lr * hr - li * hi + ur;
        const float nhi = lr * hi + li * hr + ui;
        hr = nhr; hi = nhi;
        uR[idx] = hr; uI[idx] = hi;
        idx += CHANNELS;
    }
    carR[(size_t)c * CHANNELS + ch] = hr;
    carI[(size_t)c * CHANNELS + ch] = hi;
}

// (b) sequential combine of 64 chunk carries per channel; store exclusive prefixes
__global__ __launch_bounds__(256)
void lru_carry_prefix(const float* __restrict__ carR, const float* __restrict__ carI,
                      float* __restrict__ pfxR, float* __restrict__ pfxI,
                      const float* __restrict__ nu_log, const float* __restrict__ theta_log)
{
    const int ch = blockIdx.x * blockDim.x + threadIdx.x;
    const int j  = ch & (DIM - 1);
    float lr, li; channel_lambda(nu_log, theta_log, j, lr, li);

    // Lambda = lambda^CHUNK via 6 complex squarings (CHUNK = 64 = 2^6)
    float Lr = lr, Li = li;
#pragma unroll
    for (int s = 0; s < 6; ++s) {
        const float nr = Lr * Lr - Li * Li;
        const float ni = 2.f * Lr * Li;
        Lr = nr; Li = ni;
    }

    float Pr = 0.f, Pi = 0.f;
    for (int c = 0; c < NCHUNK; ++c) {
        const size_t idx = (size_t)c * CHANNELS + ch;
        pfxR[idx] = Pr; pfxI[idx] = Pi;
        const float cr = carR[idx], ci = carI[idx];
        const float nr = Lr * Pr - Li * Pi + cr;
        const float ni = Lr * Pi + Li * Pr + ci;
        Pr = nr; Pi = ni;
    }
}

// (c) fix-up: h[t] = h_local[t] + lambda^{tau+1} * P ; write feature (M x 2D) bf16
__global__ __launch_bounds__(256)
void lru_fixup(const float* __restrict__ hR, const float* __restrict__ hI,
               const float* __restrict__ pfxR, const float* __restrict__ pfxI,
               __bf16* __restrict__ feat,
               const float* __restrict__ nu_log, const float* __restrict__ theta_log)
{
    const int ch = blockIdx.x * blockDim.x + threadIdx.x;
    const int c  = blockIdx.y;
    const int j  = ch & (DIM - 1);
    const int b  = ch >> 10;
    float lr, li; channel_lambda(nu_log, theta_log, j, lr, li);

    float fr = pfxR[(size_t)c * CHANNELS + ch];
    float fi = pfxI[(size_t)c * CHANNELS + ch];

    size_t idx = (size_t)c * CHUNK * CHANNELS + ch;
#pragma unroll 4
    for (int t = 0; t < CHUNK; ++t) {
        const float nfr = lr * fr - li * fi;   // lambda^{t+1} * P
        const float nfi = lr * fi + li * fr;
        fr = nfr; fi = nfi;
        const float vr = hR[idx] + fr;
        const float vi = hI[idx] + fi;
        const int m = (c * CHUNK + t) * BATCH + b;
        feat[(size_t)m * (2 * DIM) + j]       = (__bf16)vr;
        feat[(size_t)m * (2 * DIM) + DIM + j] = (__bf16)vi;
        idx += CHANNELS;
    }
}

// ---------- launch ----------
extern "C" void kernel_launch(void* const* d_in, const int* in_sizes, int n_in,
                              void* d_out, int out_size, void* d_ws, size_t ws_size,
                              hipStream_t stream)
{
    const float* x         = (const float*)d_in[0];
    const float* nu_log    = (const float*)d_in[1];
    const float* theta_log = (const float*)d_in[2];
    const float* gamma_log = (const float*)d_in[3];
    const float* Wi        = (const float*)d_in[4];  // (2048, 1024)
    const float* bi        = (const float*)d_in[5];  // (2048,)
    const float* Wo        = (const float*)d_in[6];  // (1024, 2048)
    const float* bo        = (const float*)d_in[7];  // (1024,)
    float* out = (float*)d_out;

    const size_t S = (size_t)MROWS * DIM;            // 16M elements
    float*  uR   = (float*)d_ws;                     // (M, D) f32 -> h_local_r
    float*  uI   = uR + S;                           // (M, D) f32 -> h_local_i
    float*  carR = uI + S;                           // (NCHUNK, CHANNELS)
    float*  carI = carR + (size_t)NCHUNK * CHANNELS;
    float*  pfxR = carI + (size_t)NCHUNK * CHANNELS;
    float*  pfxI = pfxR + (size_t)NCHUNK * CHANNELS;
    __bf16* xbf  = (__bf16*)(pfxI + (size_t)NCHUNK * CHANNELS);  // (M, D)
    __bf16* Wibf = xbf + S;                                      // (2D, D)
    __bf16* Wobf = Wibf + (size_t)2 * DIM * DIM;                 // (D, 2D)
    __bf16* feat = Wobf + (size_t)2 * DIM * DIM;                 // (M, 2D)

    // 0) pre-convert operands to bf16
    cvt_f32_bf16<<<dim3((unsigned)(S / 4 / 256)), 256, 0, stream>>>(x, xbf, (int)(S / 4));
    cvt_f32_bf16<<<dim3(2 * DIM * DIM / 4 / 256), 256, 0, stream>>>(Wi, Wibf, 2 * DIM * DIM / 4);
    cvt_f32_bf16<<<dim3(2 * DIM * DIM / 4 / 256), 256, 0, stream>>>(Wo, Wobf, 2 * DIM * DIM / 4);

    // 1) U = deinterleave(gamma * (x @ Wi^T + bi))   -> uR, uI (f32)
    {
        dim3 grid(2 * DIM / BN, MROWS / BM);         // (16, 128)
        lru_gemm_bf16<<<grid, 256, 0, stream>>>(
            xbf, /*ldA=*/DIM, Wibf, /*K=*/DIM, /*N=*/2 * DIM,
            bi, gamma_log, uR, uI, /*mode=*/0);
    }
    // 2) chunk-local scans (in place)
    {
        dim3 grid(CHANNELS / 256, NCHUNK);           // (16, 64)
        lru_chunk_scan<<<grid, 256, 0, stream>>>(uR, uI, carR, carI, nu_log, theta_log);
    }
    // 3) carry prefix combine
    lru_carry_prefix<<<dim3(CHANNELS / 256), 256, 0, stream>>>(
        carR, carI, pfxR, pfxI, nu_log, theta_log);
    // 4) fix-up -> feature (M x 2D) bf16
    {
        dim3 grid(CHANNELS / 256, NCHUNK);
        lru_fixup<<<grid, 256, 0, stream>>>(uR, uI, pfxR, pfxI, feat, nu_log, theta_log);
    }
    // 5) out = feature @ Wo^T + bo
    {
        dim3 grid(DIM / BN, MROWS / BM);             // (8, 128)
        lru_gemm_bf16<<<grid, 256, 0, stream>>>(
            feat, /*ldA=*/2 * DIM, Wobf, /*K=*/2 * DIM, /*N=*/DIM,
            bo, /*gamma_log=*/nullptr, out, /*outI=*/nullptr, /*mode=*/1);
    }
}
